// SimpleGCN_5342939316786
// MI455X (gfx1250) — matrix operand order repaired and checked by reading, not compiled, not run on hardware
//
#include <hip/hip_runtime.h>
#include <hip/hip_bf16.h>

#define N_NODES 50000
#define N_EDGES 1600000
#define F_IN    128
#define F_MID   64
#define F_OUT   2
#define N_TOT   (N_EDGES + N_NODES)   // edges + self loops

typedef __attribute__((ext_vector_type(16))) __bf16 v16bf;
typedef __attribute__((ext_vector_type(8)))  float  v8f;

// ---------------------------------------------------------------------------
// 0) init: deg=0, agg1=0, out = b2 broadcast (atomic destinations must be
//    re-initialized every call; harness does not re-poison between replays)
// ---------------------------------------------------------------------------
__global__ void gcn_init(float* __restrict__ deg, float* __restrict__ agg1,
                         float* __restrict__ out, const float* __restrict__ b2) {
    int i = blockIdx.x * blockDim.x + threadIdx.x;
    if (i < N_NODES * F_MID) agg1[i] = 0.0f;
    if (i < N_NODES)         deg[i]  = 0.0f;
    if (i < N_NODES * F_OUT) out[i]  = b2[i & 1];
}

// ---------------------------------------------------------------------------
// 1) degree via hardware f32 atomics (self loops appended: deg >= 1)
//    edge_index flat layout: [0..E) = src row, [E..2E) = dst row
// ---------------------------------------------------------------------------
__global__ void gcn_degree(const int* __restrict__ ei, float* __restrict__ deg) {
    int e = blockIdx.x * blockDim.x + threadIdx.x;
    if (e >= N_TOT) return;
    int d = (e < N_EDGES) ? ei[N_EDGES + e] : (e - N_EDGES);
    unsafeAtomicAdd(&deg[d], 1.0f);
}

// 2) dinv = rsqrt(deg), in place
__global__ void gcn_rsqrt(float* __restrict__ deg) {
    int i = blockIdx.x * blockDim.x + threadIdx.x;
    if (i < N_NODES) deg[i] = rsqrtf(deg[i]);
}

// ---------------------------------------------------------------------------
// 3) h1 = x @ W1 via v_wmma_f32_16x16x32_bf16.
//    Block = 128 thr = 4 waves; wave w computes 16x16 tile (rows 16*bid,
//    cols 16*w). 50000 % 16 == 0 -> no bounds branch, EXEC all-ones.
//    A frag (16x32 bf16): lane l<16 -> row l, elems {0..7}=K kk..kk+7,
//    {8..15}=K kk+16..kk+23; lanes 16..31 same rows, K offset +8.
//    B frag (32x16 bf16): lane l -> col l&15, K same as A frag.
// ---------------------------------------------------------------------------
__global__ void __launch_bounds__(128)
gcn_gemm1(const float* __restrict__ x, const float* __restrict__ W1,
          float* __restrict__ h1) {
    const int lane = threadIdx.x & 31;
    const int wave = threadIdx.x >> 5;
    const int r0   = blockIdx.x * 16;
    const int c0   = wave * 16;
    const int rlo  = lane & 15;
    const int hi   = lane >> 4;          // 0 or 1

    v8f acc = {};
    const float* arow = x + (r0 + rlo) * F_IN + hi * 8;
#pragma unroll
    for (int kk = 0; kk < F_IN; kk += 32) {
        v16bf A;
        const float* ap = arow + kk;
#pragma unroll
        for (int j = 0; j < 8; ++j) {
            A[j]     = (__bf16)ap[j];        // K = kk + hi*8 + j
            A[j + 8] = (__bf16)ap[j + 16];   // K = kk + hi*8 + 16 + j
        }
        v16bf B;
        const float* bp = W1 + (kk + hi * 8) * F_MID + c0 + rlo;
#pragma unroll
        for (int j = 0; j < 8; ++j) {
            B[j]     = (__bf16)bp[j * F_MID];
            B[j + 8] = (__bf16)bp[(16 + j) * F_MID];
        }

        acc = __builtin_amdgcn_wmma_f32_16x16x32_bf16(
                  false, A, false, B, (short)0, acc, false, false);
    }
    // D: n = c0 + (lane&15), m = r0 + vgpr + 8*hi
    float* o = h1 + (r0 + hi * 8) * F_MID + c0 + rlo;
#pragma unroll
    for (int v = 0; v < 8; ++v) o[v * F_MID] = acc[v];
}

// ---------------------------------------------------------------------------
// 4) scatter-add layer 1: agg1[dst,f] += h1[src,f] * dinv[src]*dinv[dst]
//    one lane per (edge, feature): coalesced 4B loads + L2-resident atomics
// ---------------------------------------------------------------------------
__global__ void gcn_scatter1(const int* __restrict__ ei,
                             const float* __restrict__ dinv,
                             const float* __restrict__ h1,
                             float* __restrict__ agg1) {
    long long idx = (long long)blockIdx.x * blockDim.x + threadIdx.x;
    long long e = idx >> 6;
    int f = (int)(idx & 63);
    if (e >= (long long)N_TOT) return;
    int s, d;
    if (e < N_EDGES) { s = ei[e]; d = ei[N_EDGES + e]; }
    else             { s = d = (int)(e - N_EDGES); }
    float nrm = dinv[s] * dinv[d];
    unsafeAtomicAdd(&agg1[d * F_MID + f], h1[s * F_MID + f] * nrm);
}

// ---------------------------------------------------------------------------
// 5) h2 = relu(agg1 + b1) @ W2   (N=2: scalar FMA, 12.8 MFLOP total)
// ---------------------------------------------------------------------------
__global__ void gcn_layer2(const float* __restrict__ agg1,
                           const float* __restrict__ b1,
                           const float* __restrict__ W2,
                           float* __restrict__ h2) {
    int i = blockIdx.x * blockDim.x + threadIdx.x;
    if (i >= N_NODES) return;
    float a0 = 0.0f, a1 = 0.0f;
#pragma unroll 8
    for (int k = 0; k < F_MID; ++k) {
        float v = fmaxf(agg1[i * F_MID + k] + b1[k], 0.0f);
        a0 = fmaf(v, W2[2 * k + 0], a0);
        a1 = fmaf(v, W2[2 * k + 1], a1);
    }
    h2[2 * i + 0] = a0;
    h2[2 * i + 1] = a1;
}

// ---------------------------------------------------------------------------
// 6) scatter-add layer 2 into d_out (pre-initialized to b2)
// ---------------------------------------------------------------------------
__global__ void gcn_scatter2(const int* __restrict__ ei,
                             const float* __restrict__ dinv,
                             const float* __restrict__ h2,
                             float* __restrict__ out) {
    int e = blockIdx.x * blockDim.x + threadIdx.x;
    if (e >= N_TOT) return;
    int s, d;
    if (e < N_EDGES) { s = ei[e]; d = ei[N_EDGES + e]; }
    else             { s = d = e - N_EDGES; }
    float nrm = dinv[s] * dinv[d];
    unsafeAtomicAdd(&out[2 * d + 0], h2[2 * s + 0] * nrm);
    unsafeAtomicAdd(&out[2 * d + 1], h2[2 * s + 1] * nrm);
}

// ---------------------------------------------------------------------------
extern "C" void kernel_launch(void* const* d_in, const int* in_sizes, int n_in,
                              void* d_out, int out_size, void* d_ws, size_t ws_size,
                              hipStream_t stream) {
    const float* x  = (const float*)d_in[0];
    const int*   ei = (const int*)d_in[1];   // harness passes integer arrays as int32
    const float* W1 = (const float*)d_in[2];
    const float* b1 = (const float*)d_in[3];
    const float* W2 = (const float*)d_in[4];
    const float* b2 = (const float*)d_in[5];
    float* out = (float*)d_out;

    // workspace layout (floats): dinv | h1 | agg1 | h2  (~26.2 MB)
    float* ws   = (float*)d_ws;
    float* dinv = ws;
    float* h1   = dinv + N_NODES;
    float* agg1 = h1   + (size_t)N_NODES * F_MID;
    float* h2   = agg1 + (size_t)N_NODES * F_MID;

    const int B = 256;

    gcn_init<<<(N_NODES * F_MID + B - 1) / B, B, 0, stream>>>(dinv, agg1, out, b2);
    gcn_degree<<<(N_TOT + B - 1) / B, B, 0, stream>>>(ei, dinv);
    gcn_rsqrt<<<(N_NODES + B - 1) / B, B, 0, stream>>>(dinv);
    gcn_gemm1<<<N_NODES / 16, 128, 0, stream>>>(x, W1, h1);
    {
        long long work = (long long)N_TOT * F_MID;
        gcn_scatter1<<<(unsigned)((work + B - 1) / B), B, 0, stream>>>(ei, dinv, h1, agg1);
    }
    gcn_layer2<<<(N_NODES + B - 1) / B, B, 0, stream>>>(agg1, b1, W2, h2);
    gcn_scatter2<<<(N_TOT + B - 1) / B, B, 0, stream>>>(ei, dinv, h2, out);
}
